// MultiHeadAttentionBlock_19250043420691
// MI455X (gfx1250) — compile-verified, matrix operand order checked by reading, not run on hardware
//
#include <hip/hip_runtime.h>
#include <hip/hip_bf16.h>

// ---------------------------------------------------------------------------
// MHA block on gfx1250 (MI455X): all GEMMs via v_wmma_f32_16x16x32_bf16,
// flash-style online softmax, one-shot bf16 conversion, 16x64 GEMM strips.
// ---------------------------------------------------------------------------

typedef __attribute__((ext_vector_type(16))) __bf16 v16bf;
typedef __attribute__((ext_vector_type(8)))  __bf16 v8bf;
typedef __attribute__((ext_vector_type(8)))  float  v8f;

static constexpr int B_ = 2;
static constexpr int S_ = 2048;
static constexpr int D_ = 512;
static constexpr int H_ = 8;
static constexpr int DK_ = 64;

__device__ __forceinline__ v8f wmma_bf16(v16bf a, v16bf b, v8f c) {
  // (neg_a, A, neg_b, B, c_mod, C, reuse_a, reuse_b)
  return __builtin_amdgcn_wmma_f32_16x16x32_bf16(false, a, false, b,
                                                 (short)0, c, false, false);
}

// reductions over a 16-lane half (rows of one C-fragment live in one half)
__device__ __forceinline__ float half_max(float v) {
  v = fmaxf(v, __shfl_xor(v, 1, 32));
  v = fmaxf(v, __shfl_xor(v, 2, 32));
  v = fmaxf(v, __shfl_xor(v, 4, 32));
  v = fmaxf(v, __shfl_xor(v, 8, 32));
  return v;
}
__device__ __forceinline__ float half_sum(float v) {
  v += __shfl_xor(v, 1, 32);
  v += __shfl_xor(v, 2, 32);
  v += __shfl_xor(v, 4, 32);
  v += __shfl_xor(v, 8, 32);
  return v;
}

// ---------------------------------------------------------------------------
// Kernel 0: one-shot fp32 -> bf16 conversion (8 elems/thread, b128 I/O).
// Removes all fp32 traffic and cvt VALU work from the GEMM hot loops.
// ---------------------------------------------------------------------------
__global__ void cvt_bf16_kernel(const float* __restrict__ src,
                                __bf16* __restrict__ dst, int n) {
  const int i = (blockIdx.x * blockDim.x + threadIdx.x) * 8;
  if (i >= n) return;
  v8f x = *(const v8f*)(src + i);
  v8bf r;
#pragma unroll
  for (int j = 0; j < 8; ++j) r[j] = (__bf16)x[j];
  *(v8bf*)(dst + i) = r;
}

// ---------------------------------------------------------------------------
// Kernel 1: Q/K/V projection, Y = X @ Wq^T, all-bf16 operands.
// One wave computes a 16x64 strip: A-fragment reused across 4 WMMAs/K-step.
// z==0 -> Qb row-major, z==1 -> Kb row-major, z==2 -> Vt transposed [b*D+n][s]
// ---------------------------------------------------------------------------
__global__ void qkv_proj_kernel(const __bf16* __restrict__ Xq,
                                const __bf16* __restrict__ Xk,
                                const __bf16* __restrict__ Xv,
                                const __bf16* __restrict__ Wqb,
                                __bf16* __restrict__ Qb,
                                __bf16* __restrict__ Kb,
                                __bf16* __restrict__ Vt) {
  const int lane = threadIdx.x;
  const int col  = lane & 15;
  const int half = lane >> 4;
  const int m0 = blockIdx.x * 16;   // row tile over B*S
  const int n0 = blockIdx.y * 64;   // 64-wide output strip over D
  const int which = blockIdx.z;

  const __bf16* src = (which == 0) ? Xq : (which == 1) ? Xk : Xv;

  v8f acc[4] = {};
#pragma unroll 2
  for (int d = 0; d < D_; d += 32) {
    // A: rows of X (lane=row, d-contiguous) — reused by 4 WMMAs
    v16bf a = *(const v16bf*)(src + (size_t)(m0 + col) * D_ + d + half * 16);
#pragma unroll
    for (int t = 0; t < 4; ++t) {
      // B: column n of B == row n of Wq (d-contiguous)
      v16bf bf = *(const v16bf*)(Wqb + (size_t)(n0 + t * 16 + col) * D_ +
                                 d + half * 16);
      acc[t] = wmma_bf16(a, bf, acc[t]);
    }
  }

  if (which < 2) {
    __bf16* dst = (which == 0) ? Qb : Kb;
#pragma unroll
    for (int t = 0; t < 4; ++t)
#pragma unroll
      for (int j = 0; j < 8; ++j) {
        const int m = m0 + j + 8 * half;
        dst[(size_t)m * D_ + n0 + t * 16 + col] = (__bf16)acc[t][j];
      }
  } else {
#pragma unroll
    for (int t = 0; t < 4; ++t)
#pragma unroll
      for (int j = 0; j < 8; ++j) {
        const int m = m0 + j + 8 * half;          // global row over B*S
        const int b = m / S_, s = m - b * S_;
        const int n = n0 + t * 16 + col;          // == h*DK + dk
        Vt[((size_t)b * D_ + n) * S_ + s] = (__bf16)acc[t][j];
      }
  }
}

// ---------------------------------------------------------------------------
// Kernel 2: flash attention. One wave per (b, h, 16-query tile).
// Per iteration: 32 keys -> 4 WMMAs for scores, masked-fill(1e-9), online
// softmax (shfl_xor row reductions), P via LDS (C-layout -> A-layout),
// 4 WMMAs for O += P @ V (B-fragments contiguous thanks to Vt transpose).
// ---------------------------------------------------------------------------
__global__ void flash_attn_kernel(const __bf16* __restrict__ Qb,
                                  const __bf16* __restrict__ Kb,
                                  const __bf16* __restrict__ Vt,
                                  const int* __restrict__ mask,
                                  __bf16* __restrict__ Xb) {
  const int lane = threadIdx.x;
  const int col  = lane & 15;
  const int half = lane >> 4;
  const int q0 = blockIdx.x * 16;
  const int h  = blockIdx.y;
  const int b  = blockIdx.z;

  __shared__ alignas(32) __bf16 pbuf[16 * 32];

  // Q A-fragments for d-chunks [0,32) and [32,64)
  const size_t qbase = ((size_t)(b * S_) + q0 + col) * D_ + h * DK_ + half * 16;
  const v16bf qa0 = *(const v16bf*)(Qb + qbase);
  const v16bf qa1 = *(const v16bf*)(Qb + qbase + 32);

  const __bf16* kb_head = Kb + (size_t)(b * S_) * D_ + h * DK_;
  const __bf16* vt_head = Vt + ((size_t)b * D_ + h * DK_) * S_;

  v8f o0 = {}, o1 = {}, o2 = {}, o3 = {};
  float m_r[8], l_r[8];
#pragma unroll
  for (int j = 0; j < 8; ++j) { m_r[j] = -1e30f; l_r[j] = 0.0f; }

  const float scale = 0.125f;  // 1/sqrt(64)

  for (int k0 = 0; k0 < S_; k0 += 32) {
    // ---- scores S = Q K^T : B-fragment lane = key column, d-contiguous ----
    const __bf16* kp0 = kb_head + (size_t)(k0 + col) * D_ + half * 16;
    const __bf16* kp1 = kb_head + (size_t)(k0 + 16 + col) * D_ + half * 16;
    v16bf kb00 = *(const v16bf*)(kp0);
    v16bf kb01 = *(const v16bf*)(kp0 + 32);
    v16bf kb10 = *(const v16bf*)(kp1);
    v16bf kb11 = *(const v16bf*)(kp1 + 32);

    if (k0 + 32 < S_) {  // uniform prefetch of next key tile (global_prefetch_b8)
      __builtin_prefetch(kb_head + (size_t)(k0 + 32 + col) * D_ + half * 16, 0, 1);
      __builtin_prefetch(vt_head + (size_t)col * S_ + k0 + 32 + half * 16, 0, 1);
    }

    v8f s0 = {}, s1 = {};
    s0 = wmma_bf16(qa0, kb00, s0);
    s0 = wmma_bf16(qa1, kb01, s0);
    s1 = wmma_bf16(qa0, kb10, s1);
    s1 = wmma_bf16(qa1, kb11, s1);

    // ---- masked fill (1e-9 where mask==0, BEFORE softmax) ----
    const int* mcol = mask + (size_t)q0 * S_ + k0 + col;
    float f0[8], f1[8];
#pragma unroll
    for (int j = 0; j < 8; ++j) {
      const size_t roff = (size_t)(j + 8 * half) * S_;
      const int mv0 = mcol[roff];
      const int mv1 = mcol[roff + 16];
      f0[j] = mv0 ? s0[j] * scale : 1e-9f;
      f1[j] = mv1 ? s1[j] * scale : 1e-9f;
    }

    // ---- online softmax: row m lives in one VGPR slot across one half ----
#pragma unroll
    for (int j = 0; j < 8; ++j) {
      const float tmax = half_max(fmaxf(f0[j], f1[j]));
      const float mnew = fmaxf(m_r[j], tmax);
      const float corr = __expf(m_r[j] - mnew);
      const float p0 = __expf(f0[j] - mnew);
      const float p1 = __expf(f1[j] - mnew);
      l_r[j] = l_r[j] * corr + half_sum(p0 + p1);
      m_r[j] = mnew;
      o0[j] *= corr; o1[j] *= corr; o2[j] *= corr; o3[j] *= corr;
      const int r = j + 8 * half;
      pbuf[r * 32 + col]      = (__bf16)p0;
      pbuf[r * 32 + col + 16] = (__bf16)p1;
    }
    __syncthreads();

    // P in A-layout: lane = query row, key-contiguous
    v16bf pa = *(const v16bf*)&pbuf[col * 32 + half * 16];

    // V B-fragments from transposed Vt: lane = d column, key-contiguous
    const __bf16* vp = vt_head + (size_t)col * S_ + k0 + half * 16;
    v16bf vb0 = *(const v16bf*)(vp);
    v16bf vb1 = *(const v16bf*)(vp + 16 * S_);
    v16bf vb2 = *(const v16bf*)(vp + 32 * S_);
    v16bf vb3 = *(const v16bf*)(vp + 48 * S_);

    o0 = wmma_bf16(pa, vb0, o0);
    o1 = wmma_bf16(pa, vb1, o1);
    o2 = wmma_bf16(pa, vb2, o2);
    o3 = wmma_bf16(pa, vb3, o3);
    __syncthreads();
  }

  // ---- normalize and store merged-head output (bf16, row-major) ----
#pragma unroll
  for (int j = 0; j < 8; ++j) {
    const int r = j + 8 * half;
    const float inv = 1.0f / l_r[j];
    const size_t rowoff = ((size_t)(b * S_) + q0 + r) * D_ + h * DK_;
    Xb[rowoff + col]      = (__bf16)(o0[j] * inv);
    Xb[rowoff + 16 + col] = (__bf16)(o1[j] * inv);
    Xb[rowoff + 32 + col] = (__bf16)(o2[j] * inv);
    Xb[rowoff + 48 + col] = (__bf16)(o3[j] * inv);
  }
}

// ---------------------------------------------------------------------------
// Kernel 3: output projection. out = X @ Wo^T, 16x64 strip/wave, fp32 result.
// ---------------------------------------------------------------------------
__global__ void out_proj_kernel(const __bf16* __restrict__ Xb,
                                const __bf16* __restrict__ Wob,
                                float* __restrict__ out) {
  const int lane = threadIdx.x;
  const int col  = lane & 15;
  const int half = lane >> 4;
  const int m0 = blockIdx.x * 16;
  const int n0 = blockIdx.y * 64;

  v8f acc[4] = {};
#pragma unroll 2
  for (int d = 0; d < D_; d += 32) {
    v16bf a = *(const v16bf*)(Xb + (size_t)(m0 + col) * D_ + d + half * 16);
#pragma unroll
    for (int t = 0; t < 4; ++t) {
      v16bf bf = *(const v16bf*)(Wob + (size_t)(n0 + t * 16 + col) * D_ +
                                 d + half * 16);
      acc[t] = wmma_bf16(a, bf, acc[t]);
    }
  }
#pragma unroll
  for (int t = 0; t < 4; ++t)
#pragma unroll
    for (int j = 0; j < 8; ++j) {
      const int m = m0 + j + 8 * half;
      out[(size_t)m * D_ + n0 + t * 16 + col] = acc[t][j];
    }
}

// ---------------------------------------------------------------------------
extern "C" void kernel_launch(void* const* d_in, const int* in_sizes, int n_in,
                              void* d_out, int out_size, void* d_ws, size_t ws_size,
                              hipStream_t stream) {
  (void)in_sizes; (void)n_in; (void)out_size; (void)ws_size;

  const float* q    = (const float*)d_in[0];
  const float* k    = (const float*)d_in[1];
  const float* v    = (const float*)d_in[2];
  const int*   mask = (const int*)d_in[3];
  const float* Wq   = (const float*)d_in[4];
  const float* Wo   = (const float*)d_in[5];
  float* out = (float*)d_out;

  char* ws = (char*)d_ws;
  const size_t elems = (size_t)B_ * S_ * D_;            // 2M elements
  const size_t welems = (size_t)D_ * D_;                // 256K elements
  __bf16* Qb  = (__bf16*)(ws + 0 * elems * 2);          // 4 MB each
  __bf16* Kb  = (__bf16*)(ws + 1 * elems * 2);
  __bf16* Vt  = (__bf16*)(ws + 2 * elems * 2);
  __bf16* Xb  = (__bf16*)(ws + 3 * elems * 2);
  __bf16* Xq  = (__bf16*)(ws + 4 * elems * 2);          // bf16 copies of inputs
  __bf16* Xk  = (__bf16*)(ws + 5 * elems * 2);
  __bf16* Xv  = (__bf16*)(ws + 6 * elems * 2);
  __bf16* Wqb = (__bf16*)(ws + 7 * elems * 2);
  __bf16* Wob = Wqb + welems;

  // --- one-shot bf16 conversions ---
  const int ethreads = (int)(elems / 8);
  const int wthreads = (int)(welems / 8);
  cvt_bf16_kernel<<<dim3(ethreads / 256), dim3(256), 0, stream>>>(q,  Xq,  (int)elems);
  cvt_bf16_kernel<<<dim3(ethreads / 256), dim3(256), 0, stream>>>(k,  Xk,  (int)elems);
  cvt_bf16_kernel<<<dim3(ethreads / 256), dim3(256), 0, stream>>>(v,  Xv,  (int)elems);
  cvt_bf16_kernel<<<dim3(wthreads / 256), dim3(256), 0, stream>>>(Wq, Wqb, (int)welems);
  cvt_bf16_kernel<<<dim3(wthreads / 256), dim3(256), 0, stream>>>(Wo, Wob, (int)welems);

  dim3 g1(B_ * S_ / 16, D_ / 64, 3);                    // 256 x 8 x 3
  qkv_proj_kernel<<<g1, dim3(32), 0, stream>>>(Xq, Xk, Xv, Wqb, Qb, Kb, Vt);

  dim3 g2(S_ / 16, H_, B_);                             // 128 x 8 x 2
  flash_attn_kernel<<<g2, dim3(32), 0, stream>>>(Qb, Kb, Vt, mask, Xb);

  dim3 g3(B_ * S_ / 16, D_ / 64, 1);                    // 256 x 8
  out_proj_kernel<<<g3, dim3(32), 0, stream>>>(Xb, Wob, out);
}
